// AttentionHead_19275813224653
// MI455X (gfx1250) — matrix operand order repaired and checked by reading, not compiled, not run on hardware
//
#include <hip/hip_runtime.h>
#include <hip/hip_bf16.h>

// ---------------------------------------------------------------------------
// AttentionHead on gfx1250 (CDNA5): bf16 WMMA flash attention.
//   B=16, S=2048, W=768, H=64.
// Pipeline:
//   k0: repack Wq/Wk/Wv into WMMA B-matrix fragment layout (bf16),
//       grouped per k-step for clause-friendly streaming
//   k1: QKV projection via v_wmma_f32_16x16x32_bf16, 2 waves/block (6 output
//       tiles each -> low VGPR pressure, 6 B-frags in flight, 4096 waves for
//       the HBM-bound input stream); writes Q,K [B,S,64] bf16 and V
//       transposed [B,64,S] bf16 (contiguous 32B attention B-fragments)
//   k2: flash attention, one wave per 16-query tile, 64-key chunks, exp2-
//       domain online softmax; row sums via WMMA against a ones matrix; P
//       staged C-layout -> A-layout through 2KB LDS (in-order DS, no barrier);
//       global_prefetch of next chunk's K/V.
// Workspace: Qb(4MB) + Kb(4MB) + Vt(4MB) + wfrag(288KB) ~= 12.3 MB.
// ---------------------------------------------------------------------------

typedef __bf16 bf16_t;
typedef bf16_t v16bf __attribute__((ext_vector_type(16)));
typedef bf16_t v8bf  __attribute__((ext_vector_type(8)));
typedef float  v8f   __attribute__((ext_vector_type(8)));

#define BB 16
#define SS 2048
#define WW 768
#define HH 64
#define QT 16   // queries per wave tile
#define KC 64   // keys per chunk

// 1/sqrt(64) * log2(e): scores scaled straight into the exp2 domain.
#define SCALE_LOG2E 0.1803368801111204f

static __device__ __forceinline__ float fast_exp2(float x) {
    return __builtin_amdgcn_exp2f(x);    // v_exp_f32 (native base-2)
}

static __device__ __forceinline__ v8f wmma_bf16(v16bf a, v16bf b, v8f c) {
    // D(16x16 f32) = A(16x32 bf16) * B(32x16 bf16) + C
    return __builtin_amdgcn_wmma_f32_16x16x32_bf16(false, a, false, b,
                                                   (short)0, c, false, false);
}

static __device__ __forceinline__ v8f v8f_zero() {
    v8f z = {0.f, 0.f, 0.f, 0.f, 0.f, 0.f, 0.f, 0.f};
    return z;
}

// ---------------------------------------------------------------------------
// Kernel 0: repack weights [768][64] f32 -> bf16 B-fragments.
// Block id = ks*12 + w*4 + n. Fragment: lane holds column N = n*16+(lane&15),
// element e holds K = ks*32 + e + 16*(lane>>4).   (contiguous k order, same as the A fragments)
// Layout [ks][w][n][lane][16] so one k-step's 12 frags are 12KB contiguous.
// ---------------------------------------------------------------------------
__global__ void __launch_bounds__(32)
repack_weights(const float* __restrict__ Wq, const float* __restrict__ Wk,
               const float* __restrict__ Wv, bf16_t* __restrict__ frag) {
    int blk = blockIdx.x;          // 0..287
    int ks  = blk / 12;
    int rem = blk % 12;
    int w   = rem / 4;
    int n   = rem % 4;
    int lane = threadIdx.x;
    int lo = lane & 15, hi = lane >> 4;
    const float* Wsrc = (w == 0) ? Wq : (w == 1) ? Wk : Wv;

    v16bf f;
#pragma unroll
    for (int e = 0; e < 16; ++e) {
        int row = ks * 32 + e + 16 * hi;   // K (input-feature) index
        int col = n * 16 + lo;             // N (head) index
        f[e] = (bf16_t)Wsrc[row * HH + col];
    }
    bf16_t* dst = frag + ((size_t)blk * 32 + lane) * 16;
    *(v16bf*)dst = f;
}

// ---------------------------------------------------------------------------
// Kernel 1: QKV projection. 2 waves per block; each wave owns one 16-row tile
// and 6 of the 12 output subtiles (wave0: Q n0-3 + K n0-1; wave1: K n2-3 +
// V n0-3). A-fragment (16x32), same contiguous k order as B: lane(lo,hi) element e -> row=lo,
// K = e + 16*hi  (+ 32*kstep).
// ---------------------------------------------------------------------------
__global__ void __launch_bounds__(64)
qkv_proj(const float* __restrict__ in,   // [B,S,W]
         const float* __restrict__ bq, const float* __restrict__ bk,
         const float* __restrict__ bv,
         const bf16_t* __restrict__ frag,
         bf16_t* __restrict__ Qb,        // [B,S,H] bf16
         bf16_t* __restrict__ Kb,        // [B,S,H] bf16
         bf16_t* __restrict__ Vt) {      // [B,H,S] bf16 (transposed)
    int t = blockIdx.x;                  // 0..2047 (16-row tile index)
    int b = t >> 7;                      // S/QT = 128 tiles per batch
    int qbase = (t & 127) * QT;
    int lane = threadIdx.x & 31;
    int w2   = threadIdx.x >> 5;         // wave id: 0 or 1
    int lo = lane & 15, hi = lane >> 4;

    v8f acc[6];
#pragma unroll
    for (int i = 0; i < 6; ++i) acc[i] = v8f_zero();

    const float* arow = in + (size_t)(b * SS + qbase + lo) * WW;
    const bf16_t* fraghalf = frag + (size_t)w2 * 6 * 512 + (size_t)lane * 16;

    for (int ks = 0; ks < WW / 32; ++ks) {
        // Prefetch next k-step's slice of this row (stream from HBM).
        if (ks + 1 < WW / 32)
            __builtin_prefetch(arow + (ks + 1) * 32, 0, 3);

        // bf16 A-fragment from fp32 input (two 32B contiguous runs).
        const float* p0 = arow + ks * 32 + 16 * hi;
        const float* p1 = arow + ks * 32 + 8 + 16 * hi;
        v16bf a;
#pragma unroll
        for (int e = 0; e < 8; ++e) {
            a[e]     = (bf16_t)p0[e];
            a[e + 8] = (bf16_t)p1[e];
        }
        // Preload this wave's 6 B-fragments (one clause), then 6 WMMAs with
        // progressive waits. ~120 live VGPRs -> allocator keeps all 6.
        const bf16_t* kbase = fraghalf + (size_t)ks * 12 * 512;
        v16bf bfr[6];
#pragma unroll
        for (int i = 0; i < 6; ++i)
            bfr[i] = *(const v16bf*)(kbase + (size_t)i * 512);
#pragma unroll
        for (int i = 0; i < 6; ++i)
            acc[i] = wmma_bf16(a, bfr[i], acc[i]);
    }

    // Epilogue: bias + store. C layout: VGPR r, lane(lo,hi) -> (M=r+8hi, N=lo).
    const float* biases[3] = {bq, bk, bv};
#pragma unroll
    for (int i = 0; i < 6; ++i) {
        int gi = w2 * 6 + i;             // global subtile 0..11
        int w = gi >> 2;                 // 0=Q 1=K 2=V
        int n = gi & 3;
        float bias = biases[w][n * 16 + lo];
#pragma unroll
        for (int r = 0; r < 8; ++r) {
            float v = acc[i][r] + bias;
            int m = r + 8 * hi;
            int col = n * 16 + lo;
            size_t srow = (size_t)(b * SS + qbase + m);
            if (w == 0)      Qb[srow * HH + col] = (bf16_t)v;
            else if (w == 1) Kb[srow * HH + col] = (bf16_t)v;
            else             Vt[((size_t)b * HH + col) * SS + qbase + m] = (bf16_t)v;
        }
    }
}

// ---------------------------------------------------------------------------
// Kernel 2: flash attention. One wave per 16-query tile; 2048 keys in chunks
// of 64. Per chunk: QK^T 8 WMMAs, rowsum 2 WMMAs (P x ones), PV 8 WMMAs.
// exp2-domain online softmax; masked keys use -1e30 (never NaN: key 0 is
// always valid so the running max is finite after chunk 0).
// ---------------------------------------------------------------------------
__global__ void __launch_bounds__(32)
attn_kernel(const bf16_t* __restrict__ Qb, const bf16_t* __restrict__ Kb,
            const bf16_t* __restrict__ Vt, const int* __restrict__ mask,
            float* __restrict__ out) {
    __shared__ __align__(32) bf16_t Pl[QT * KC];   // 16x64 probs (2KB)

    int t = blockIdx.x;
    int b = t >> 7;
    int qbase = (t & 127) * QT;
    int lane = threadIdx.x;
    int lo = lane & 15, hi = lane >> 4;

    // Q A-fragments for head dims 0..31 and 32..63.
    const bf16_t* qrow = Qb + (size_t)(b * SS + qbase + lo) * HH;
    v16bf aq0, aq1;
    {
        v8bf x0 = *(const v8bf*)(qrow + 16 * hi);
        v8bf x1 = *(const v8bf*)(qrow + 8 + 16 * hi);
        v8bf x2 = *(const v8bf*)(qrow + 32 + 16 * hi);
        v8bf x3 = *(const v8bf*)(qrow + 40 + 16 * hi);
#pragma unroll
        for (int e = 0; e < 8; ++e) {
            aq0[e] = x0[e]; aq0[e + 8] = x1[e];
            aq1[e] = x2[e]; aq1[e + 8] = x3[e];
        }
    }

    // Ones B-matrix for WMMA row sums.
    v16bf ones;
#pragma unroll
    for (int e = 0; e < 16; ++e) ones[e] = (bf16_t)1.0f;

    float mrow[8], lrow[8];
    v8f accO[4];
#pragma unroll
    for (int r = 0; r < 8; ++r) { mrow[r] = -1e30f; lrow[r] = 0.f; }
#pragma unroll
    for (int n = 0; n < 4; ++n) accO[n] = v8f_zero();

    const int* maskb = mask + (size_t)b * SS;

    for (int kb = 0; kb < SS; kb += KC) {
        // ---- prefetch next chunk's K (8KB contiguous) and V (64 lines) ----
        if (kb + KC < SS) {
            const char* nk =
                (const char*)(Kb + (size_t)(b * SS + kb + KC) * HH);
            __builtin_prefetch(nk + lane * 256, 0, 3);
            __builtin_prefetch(nk + lane * 256 + 128, 0, 3);
            const bf16_t* nv = Vt + (size_t)(b * HH) * SS + kb + KC;
            __builtin_prefetch(nv + (size_t)lane * SS, 0, 3);
            __builtin_prefetch(nv + (size_t)(lane + 32) * SS, 0, 3);
        }

        // ---- scores: four 16x16 n-subtiles (keys kb+16j+lo), k=64 ----
        v8f s[4];
#pragma unroll
        for (int j = 0; j < 4; ++j) {
            const bf16_t* kp =
                Kb + (size_t)(b * SS + kb + 16 * j + lo) * HH + 16 * hi;
            v16bf b0 = *(const v16bf*)kp;          // head dims 0..31
            v16bf b1 = *(const v16bf*)(kp + 32);   // head dims 32..63
            v8f tj = wmma_bf16(aq0, b0, v8f_zero());
            s[j]   = wmma_bf16(aq1, b1, tj);
        }

        // ---- mask + scale into exp2 domain ----
        int mv[4];
#pragma unroll
        for (int j = 0; j < 4; ++j) mv[j] = maskb[kb + 16 * j + lo];
#pragma unroll
        for (int j = 0; j < 4; ++j)
#pragma unroll
            for (int r = 0; r < 8; ++r)
                s[j][r] = mv[j] ? s[j][r] * SCALE_LOG2E : -1e30f;

        // ---- online softmax: per-row max (rows r+8hi live on one half) ----
        float a_r[8];
#pragma unroll
        for (int r = 0; r < 8; ++r) {
            float cm = fmaxf(fmaxf(s[0][r], s[1][r]), fmaxf(s[2][r], s[3][r]));
#pragma unroll
            for (int d = 1; d < 16; d <<= 1) cm = fmaxf(cm, __shfl_xor(cm, d));
            float mn = fmaxf(mrow[r], cm);
            a_r[r] = fast_exp2(mrow[r] - mn);
            mrow[r] = mn;
            // probabilities: C-layout (M=r+8hi, N=16j+lo) -> LDS[M][col]
#pragma unroll
            for (int j = 0; j < 4; ++j) {
                float p = fast_exp2(s[j][r] - mn);
                Pl[(r + 8 * hi) * KC + 16 * j + lo] = (bf16_t)p;
            }
        }
        // Single wave: DS ops are issued in order, so the loads below see the
        // stores above. Explicit split-counter wait + compiler barrier keeps
        // the IR from migrating LDS ops across this point.
        asm volatile("s_wait_dscnt 0" ::: "memory");

        // ---- P A-fragments from LDS (row=lo, K=e+16hi) ----
        v16bf ap0, ap1;
        {
            const bf16_t* pr = &Pl[lo * KC + 16 * hi];
            v8bf y0 = *(const v8bf*)pr;            // keys 16hi+0..7
            v8bf y1 = *(const v8bf*)(pr + 8);      // keys 16hi+8..15
            v8bf y2 = *(const v8bf*)(pr + 32);     // keys 32+16hi+0..7
            v8bf y3 = *(const v8bf*)(pr + 40);     // keys 32+16hi+8..15
#pragma unroll
            for (int e = 0; e < 8; ++e) {
                ap0[e] = y0[e]; ap0[e + 8] = y1[e];
                ap1[e] = y2[e]; ap1[e + 8] = y3[e];
            }
        }

        // ---- row sums via WMMA: rs(M,N) = sum_k P[M,k] for every N ----
        v8f rs = wmma_bf16(ap0, ones, v8f_zero());
        rs     = wmma_bf16(ap1, ones, rs);
#pragma unroll
        for (int r = 0; r < 8; ++r) lrow[r] = lrow[r] * a_r[r] + rs[r];

        // ---- rescale running accumulator, then accO += P * V ----
#pragma unroll
        for (int n = 0; n < 4; ++n)
#pragma unroll
            for (int r = 0; r < 8; ++r) accO[n][r] *= a_r[r];
#pragma unroll
        for (int n = 0; n < 4; ++n) {
            const bf16_t* vp =
                Vt + (size_t)(b * HH + n * 16 + lo) * SS + kb + 16 * hi;
            v16bf bv0 = *(const v16bf*)vp;          // keys 0..31 of chunk
            v16bf bv1 = *(const v16bf*)(vp + 32);   // keys 32..63 of chunk
            accO[n] = wmma_bf16(ap0, bv0, accO[n]);
            accO[n] = wmma_bf16(ap1, bv1, accO[n]);
        }
        // WAR to Pl next iteration is safe: DS pipeline is in-order per wave.
    }

    // ---- epilogue: divide by row sums, store fp32 [B,S,64] ----
#pragma unroll
    for (int r = 0; r < 8; ++r) {
        float inv = 1.0f / lrow[r];
        int m = r + 8 * hi;
        size_t base = (size_t)(b * SS + qbase + m) * HH;
#pragma unroll
        for (int n = 0; n < 4; ++n)
            out[base + n * 16 + lo] = accO[n][r] * inv;
    }
}

// ---------------------------------------------------------------------------
extern "C" void kernel_launch(void* const* d_in, const int* in_sizes, int n_in,
                              void* d_out, int out_size, void* d_ws, size_t ws_size,
                              hipStream_t stream) {
    const float* inp = (const float*)d_in[0];
    const int*  mask = (const int*)d_in[1];
    const float* Wq  = (const float*)d_in[2];
    const float* bq  = (const float*)d_in[3];
    const float* Wk  = (const float*)d_in[4];
    const float* bk  = (const float*)d_in[5];
    const float* Wv  = (const float*)d_in[6];
    const float* bv  = (const float*)d_in[7];
    float* out = (float*)d_out;

    char* ws = (char*)d_ws;
    const size_t qkvBytes = (size_t)BB * SS * HH * sizeof(bf16_t);  // 4 MB each
    bf16_t* Qb   = (bf16_t*)(ws);
    bf16_t* Kb   = (bf16_t*)(ws + qkvBytes);
    bf16_t* Vt   = (bf16_t*)(ws + 2 * qkvBytes);
    bf16_t* frag = (bf16_t*)(ws + 3 * qkvBytes);   // 288 KB

    repack_weights<<<3 * 24 * 4, 32, 0, stream>>>(Wq, Wk, Wv, frag);
    qkv_proj<<<BB * SS / QT, 64, 0, stream>>>(inp, bq, bk, bv, frag, Qb, Kb, Vt);
    attn_kernel<<<BB * SS / QT, 32, 0, stream>>>(Qb, Kb, Vt, mask, out);
}
